// GpvaeEncoder_32384053412333
// MI455X (gfx1250) — matrix operand order repaired and checked by reading, not compiled, not run on hardware
//
#include <hip/hip_runtime.h>
#include <math.h>

// B=8, T=256, D_IN=64, H=128, Z=64, K=24, PAD=12
// Outputs: mean (B,Z,T)=131072 f32, scale_tril (B,Z,T,T)=33554432 f32, concatenated.

typedef __attribute__((ext_vector_type(16))) _Float16 v16h;
typedef __attribute__((ext_vector_type(8)))  float    v8f;
typedef __attribute__((ext_vector_type(4)))  float    v4f;

#define BB 8
#define TT 256
#define DIN 64
#define HH 128
#define ZZ 64
#define KK 24

// workspace byte offsets
#define OFF_WC 0u          // conv  B-frags: 48*8*512 f16  = 393216 B
#define OFF_WL 393216u     // lin   B-frags:  4*8*512 f16  =  32768 B
#define OFF_WM 425984u     // mu    B-frags:  4*4*512 f16  =  16384 B
#define OFF_WV 442368u     // lv    B-frags:  4*8*512 f16  =  32768 B
#define OFF_D  524288u     // d = sigmoid(lv even ch): B*Z*T f32 = 524288 B
#define OFF_S  1048576u    // s = sigmoid(lv odd  ch): B*Z*T f32

// ---- B-fragment element mapping (wave32 WMMA f16, 32x16 B tile) ----
// lane in [0,16): rows K=j (j=0..15);  lane in [16,32): rows K=16+j. col = lane&15.
// frag storage: frag[((kstep*NT+nt)*32 + lane)*16 + j]

__global__ void gpvae_pack_conv(const float* __restrict__ conv_w, _Float16* __restrict__ dst) {
    int e = blockIdx.x * blockDim.x + threadIdx.x;   // 48*8*512 = 196608
    if (e >= 48 * 8 * 512) return;
    int j    = e & 15;
    int lane = (e >> 4) & 31;
    int blk  = e >> 9;
    int nt   = blk & 7;
    int kstep= blk >> 3;
    int krow = (lane & 16) + j;
    int kk   = kstep * 32 + krow;        // im2col index: kk = tap*64 + ci
    int h    = nt * 16 + (lane & 15);
    int ci   = kk & 63;
    int tap  = kk >> 6;
    dst[e] = (_Float16)conv_w[(h * DIN + ci) * KK + tap];
}

__global__ void gpvae_pack_dense(const float* __restrict__ w, _Float16* __restrict__ dst, int NT) {
    int e = blockIdx.x * blockDim.x + threadIdx.x;   // 4*NT*512
    if (e >= 4 * NT * 512) return;
    int j    = e & 15;
    int lane = (e >> 4) & 31;
    int blk  = e >> 9;
    int nt   = blk % NT;
    int kstep= blk / NT;
    int kk   = kstep * 32 + (lane & 16) + j;         // K index (input feature)
    int n    = nt * 16 + (lane & 15);                // output feature
    dst[e] = (_Float16)w[n * HH + kk];               // w is (N, 128), out = h @ w.T
}

__device__ __forceinline__ int a_koff(int j, int lane) {
    // A-fragment (16x32 f16): per-lane K offset for element j (contiguous k = 16*(lane/16) + j, as in the B packing)
    return j + ((lane & 16) ? 16 : 0);
}

__global__ void __launch_bounds__(32)
gpvae_stage1(const float* __restrict__ x,
             const float* __restrict__ conv_b, const float* __restrict__ lin_b,
             const float* __restrict__ mu_b,   const float* __restrict__ logvar_b,
             const _Float16* __restrict__ Wc,  const _Float16* __restrict__ Wl,
             const _Float16* __restrict__ Wm,  const _Float16* __restrict__ Wv,
             float* __restrict__ mean, float* __restrict__ darr, float* __restrict__ sarr) {
    __shared__ float ldsX[39 * DIN];      // padded x slab: rows t0-12 .. t0+26
    __shared__ float htile[16 * HH];      // 16 x 128 activation staging

    const int lane = threadIdx.x & 31;
    const int b    = blockIdx.x >> 4;
    const int t0   = (blockIdx.x & 15) << 4;
    const int M    = lane & 15;
    const int mrow = ((lane >> 4) << 3);  // +i gives output row M for C/D vgpr i

    // ---- stage x slab into LDS (zero-padded) ----
    for (int idx = lane; idx < 39 * DIN; idx += 32) {
        int row = idx >> 6, ci = idx & 63;
        int t = t0 - 12 + row;
        float v = 0.f;
        if (t >= 0 && t < TT) v = x[((size_t)b * TT + t) * DIN + ci];
        ldsX[idx] = v;
    }
    __syncthreads();

    // ---- conv as GEMM: (16 x 1536) @ (1536 x 128), 48 K-steps ----
    v8f cacc[8] = {};
    for (int ks = 0; ks < 48; ++ks) {
        v16h afrag;
        #pragma unroll
        for (int j = 0; j < 16; ++j) {
            int kk = ks * 32 + a_koff(j, lane);
            int tap = kk >> 6, ci = kk & 63;
            afrag[j] = (_Float16)ldsX[(M + tap) * DIN + ci];
        }
        const v16h* wb = (const v16h*)Wc + ((size_t)(ks * 8) << 5) + lane;
        #pragma unroll
        for (int nt = 0; nt < 8; ++nt) {
            v16h bfrag = wb[nt << 5];
            cacc[nt] = __builtin_amdgcn_wmma_f32_16x16x32_f16(
                false, afrag, false, bfrag, (short)0, cacc[nt], false, false);
        }
    }
    // hconv + bias -> LDS
    #pragma unroll
    for (int nt = 0; nt < 8; ++nt) {
        int n = nt * 16 + (lane & 15);
        float bias = conv_b[n];
        #pragma unroll
        for (int i = 0; i < 8; ++i) htile[(mrow + i) * HH + n] = cacc[nt][i] + bias;
    }
    __syncthreads();

    // ---- h = relu(hconv @ lin_w.T + lin_b): K=128 (4 steps), N=128 ----
    v8f cl[8] = {};
    #pragma unroll
    for (int ks = 0; ks < 4; ++ks) {
        v16h afrag;
        #pragma unroll
        for (int j = 0; j < 16; ++j)
            afrag[j] = (_Float16)htile[M * HH + ks * 32 + a_koff(j, lane)];
        const v16h* wb = (const v16h*)Wl + ((size_t)(ks * 8) << 5) + lane;
        #pragma unroll
        for (int nt = 0; nt < 8; ++nt) {
            v16h bfrag = wb[nt << 5];
            cl[nt] = __builtin_amdgcn_wmma_f32_16x16x32_f16(
                false, afrag, false, bfrag, (short)0, cl[nt], false, false);
        }
    }
    __syncthreads();
    #pragma unroll
    for (int nt = 0; nt < 8; ++nt) {
        int n = nt * 16 + (lane & 15);
        float bias = lin_b[n];
        #pragma unroll
        for (int i = 0; i < 8; ++i)
            htile[(mrow + i) * HH + n] = fmaxf(cl[nt][i] + bias, 0.f);
    }
    __syncthreads();

    // ---- mu (N=64) and logvar (N=128) from h, shared A-fragments ----
    v8f cm[4] = {};
    v8f cv[8] = {};
    #pragma unroll
    for (int ks = 0; ks < 4; ++ks) {
        v16h afrag;
        #pragma unroll
        for (int j = 0; j < 16; ++j)
            afrag[j] = (_Float16)htile[M * HH + ks * 32 + a_koff(j, lane)];
        const v16h* wbm = (const v16h*)Wm + ((size_t)(ks * 4) << 5) + lane;
        const v16h* wbv = (const v16h*)Wv + ((size_t)(ks * 8) << 5) + lane;
        #pragma unroll
        for (int nt = 0; nt < 4; ++nt) {
            v16h bfrag = wbm[nt << 5];
            cm[nt] = __builtin_amdgcn_wmma_f32_16x16x32_f16(
                false, afrag, false, bfrag, (short)0, cm[nt], false, false);
        }
        #pragma unroll
        for (int nt = 0; nt < 8; ++nt) {
            v16h bfrag = wbv[nt << 5];
            cv[nt] = __builtin_amdgcn_wmma_f32_16x16x32_f16(
                false, afrag, false, bfrag, (short)0, cv[nt], false, false);
        }
    }

    // ---- mean[b,z,t] = mu[b,t,z] ----
    #pragma unroll
    for (int nt = 0; nt < 4; ++nt) {
        int z = nt * 16 + (lane & 15);
        float bias = mu_b[z];
        #pragma unroll
        for (int i = 0; i < 8; ++i) {
            int t = t0 + mrow + i;
            mean[((size_t)b * ZZ + z) * TT + t] = cm[nt][i] + bias;
        }
    }
    // ---- d/s = sigmoid(logvar) split by channel parity ----
    #pragma unroll
    for (int nt = 0; nt < 8; ++nt) {
        int ch = nt * 16 + (lane & 15);
        float bias = logvar_b[ch];
        int z = ch >> 1;
        #pragma unroll
        for (int i = 0; i < 8; ++i) {
            int t = t0 + mrow + i;
            float sg = 1.f / (1.f + expf(-(cv[nt][i] + bias)));
            size_t o = ((size_t)b * ZZ + z) * TT + t;
            if (ch & 1) sarr[o] = sg; else darr[o] = sg;
        }
    }
}

// ---- stage 2: scale_tril rows. One wave per row; 4 cols/lane, 128-col chunks.
// scale_tril[r,c] = (1/a_r) * prod_{k=c}^{r-1} (-s_k/a_k) for c<=r, else 0.
// Computed as a backward running product: local 4-elem suffix products +
// one cross-lane suffix scan per chunk, b128 non-temporal stores.
__global__ void __launch_bounds__(256)
gpvae_stage2(const float* __restrict__ darr, const float* __restrict__ sarr,
             float* __restrict__ out) {
    int wid  = (blockIdx.x * blockDim.x + threadIdx.x) >> 5;  // global row id
    int lane = threadIdx.x & 31;
    int bz   = wid >> 8;          // (b*Z+z)
    int r    = wid & 255;         // row of scale_tril
    const float* dA = darr + (size_t)bz * TT;
    const float* sA = sarr + (size_t)bz * TT;
    float* outRow = out + (size_t)BB * ZZ * TT + (size_t)wid * TT;

    float carry = 1.f;
    #pragma unroll
    for (int cb = 1; cb >= 0; --cb) {
        int c0 = cb * 128 + lane * 4;
        v4f dv = *(const v4f*)(dA + c0);
        v4f sv = *(const v4f*)(sA + c0);
        float f[4];
        #pragma unroll
        for (int i = 0; i < 4; ++i) {
            int c = c0 + i;
            float a = 1.f + dv[i];
            f[i] = (c < r) ? (-sv[i] / a) : ((c == r) ? 1.f / a : 1.f);
        }
        // local suffix products within the lane's 4 columns
        float p[4];
        p[3] = f[3];
        p[2] = f[2] * p[3];
        p[1] = f[1] * p[2];
        p[0] = f[0] * p[1];
        // cross-lane inclusive suffix-product scan of lane totals (Kogge-Stone)
        float I = p[0];
        #pragma unroll
        for (int off = 1; off < 32; off <<= 1) {
            float o = __shfl(I, lane + off, 32);
            if (lane + off < 32) I *= o;
        }
        float E = __shfl(I, lane + 1, 32);   // exclusive suffix (lanes above)
        if (lane == 31) E = 1.f;
        float ce = carry * E;
        v4f ov;
        #pragma unroll
        for (int i = 0; i < 4; ++i) {
            float v = ce * p[i];
            int c = c0 + i;
            if (c > r) v = 0.f;
            if (!__builtin_isfinite(v)) v = 0.f;
            ov[i] = v;
        }
        __builtin_nontemporal_store(ov, (v4f*)(outRow + c0));
        carry *= __shfl(I, 0, 32);           // whole-chunk product
    }
}

extern "C" void kernel_launch(void* const* d_in, const int* in_sizes, int n_in,
                              void* d_out, int out_size, void* d_ws, size_t ws_size,
                              hipStream_t stream) {
    const float* x        = (const float*)d_in[0];
    const float* conv_w   = (const float*)d_in[1];
    const float* conv_b   = (const float*)d_in[2];
    const float* lin_w    = (const float*)d_in[3];
    const float* lin_b    = (const float*)d_in[4];
    const float* mu_w     = (const float*)d_in[5];
    const float* mu_b     = (const float*)d_in[6];
    const float* logvar_w = (const float*)d_in[7];
    const float* logvar_b = (const float*)d_in[8];

    char* ws = (char*)d_ws;
    _Float16* Wc = (_Float16*)(ws + OFF_WC);
    _Float16* Wl = (_Float16*)(ws + OFF_WL);
    _Float16* Wm = (_Float16*)(ws + OFF_WM);
    _Float16* Wv = (_Float16*)(ws + OFF_WV);
    float* darr  = (float*)(ws + OFF_D);
    float* sarr  = (float*)(ws + OFF_S);
    float* out   = (float*)d_out;

    gpvae_pack_conv<<<768, 256, 0, stream>>>(conv_w, Wc);
    gpvae_pack_dense<<<64, 256, 0, stream>>>(lin_w, Wl, 8);
    gpvae_pack_dense<<<32, 256, 0, stream>>>(mu_w, Wm, 4);
    gpvae_pack_dense<<<64, 256, 0, stream>>>(logvar_w, Wv, 8);

    gpvae_stage1<<<BB * (TT / 16), 32, 0, stream>>>(
        x, conv_b, lin_b, mu_b, logvar_b, Wc, Wl, Wm, Wv, out, darr, sarr);

    // B*Z*T rows, one wave32 per row, 8 waves per block
    gpvae_stage2<<<(BB * ZZ * TT) / 8, 256, 0, stream>>>(darr, sarr, out);
}
